// VGG_38362647888502
// MI455X (gfx1250) — matrix operand order repaired and checked by reading, not compiled, not run on hardware
//
#include <hip/hip_runtime.h>

// ---------------------------------------------------------------------------
// Types for CDNA5 WMMA (wave32): 16x16x32 bf16 -> f32
// ---------------------------------------------------------------------------
typedef __attribute__((ext_vector_type(16))) __bf16 v16bf;
typedef __attribute__((ext_vector_type(8)))  __bf16 v8bf;
typedef __attribute__((ext_vector_type(8)))  float  v8f;

union FragAB { v16bf f; v8bf h[2]; };

__device__ __forceinline__ unsigned short f2bf_us(float f) {
    unsigned u = __float_as_uint(f);
    return (unsigned short)((u + 0x7fffu + ((u >> 16) & 1u)) >> 16);
}

// Problem constants
#define BATCH   32
#define CIN0    512
#define HW      28
#define PIX     784            // 28*28
#define M_TOT   25088          // 32*784
#define NC      200
#define NPAD3   256

// ---------------------------------------------------------------------------
// avg_pool2d(x,3,s=1,p=1)/9  :  NCHW f32  ->  NHWC bf16 [M_TOT][512]
// ---------------------------------------------------------------------------
__global__ void k_avgpool(const float* __restrict__ x, unsigned short* __restrict__ feat) {
    int g = blockIdx.x * blockDim.x + threadIdx.x;
    if (g >= M_TOT * CIN0) return;
    int c = g & (CIN0 - 1);
    int m = g >> 9;
    int b = m / PIX; int p = m - b * PIX;
    int h = p / HW;  int w = p - h * HW;
    const float* xb = x + ((size_t)(b * CIN0 + c)) * PIX;
    float s = 0.f;
    #pragma unroll
    for (int dy = -1; dy <= 1; ++dy) {
        int hh = h + dy; if ((unsigned)hh >= (unsigned)HW) continue;
        #pragma unroll
        for (int dx = -1; dx <= 1; ++dx) {
            int ww = w + dx; if ((unsigned)ww >= (unsigned)HW) continue;
            s += xb[hh * HW + ww];
        }
    }
    feat[(size_t)m * CIN0 + c] = f2bf_us(s * (1.f / 9.f));
}

// ---------------------------------------------------------------------------
// Pack 3x3 weights [O][I][3][3] f32 -> [tap][I][O] bf16  (K-major, N contig)
// ---------------------------------------------------------------------------
__global__ void k_pack3x3(const float* __restrict__ w, unsigned short* __restrict__ wp,
                          int Cin, int O, int total) {
    int idx = blockIdx.x * blockDim.x + threadIdx.x;
    if (idx >= total) return;
    int o = idx % O;
    int r = idx / O;
    int i = r % Cin;
    int t = r / Cin;
    wp[idx] = f2bf_us(w[((size_t)(o * Cin + i)) * 9 + t]);
}

// Pack 1x1 weights [200][1024] f32 -> [1024][256] bf16, zero pad O>=200
__global__ void k_pack1x1(const float* __restrict__ w, unsigned short* __restrict__ wp) {
    int idx = blockIdx.x * blockDim.x + threadIdx.x;   // 1024*256
    if (idx >= 1024 * NPAD3) return;
    int o = idx & (NPAD3 - 1);
    int i = idx >> 8;
    float v = (o < NC) ? w[(size_t)o * 1024 + i] : 0.f;
    wp[idx] = f2bf_us(v);
}

// ---------------------------------------------------------------------------
// Implicit-GEMM conv:  OUT[M_TOT][Npad] = im2col(A) * Wp + bias
//   A   : NHWC bf16 [M_TOT][Cin]
//   Wp  : bf16 [taps][Cin][Npad]   (K-major, N contiguous)
//   taps=9 -> 3x3 pad 1 (tap folded into K loop); taps=1 -> 1x1
//   Block tile 128x128, 8 waves (wave32) in 4(M)x2(N); each wave 32x64 =
//   2x4 WMMA accumulators -> 8 v_wmma per K-step.
//   B tile staged with global_load_async_to_lds_b128 (ASYNCcnt),
//   A tile staged via regular loads + ds stores (zero-fill for 3x3 padding).
//   Tap/K decomposition uses shifts only (Cin/32 is a power of two).
// ---------------------------------------------------------------------------
__global__ __launch_bounds__(256)
void k_gemm(const unsigned short* __restrict__ A,
            const unsigned short* __restrict__ Wp,
            const float* __restrict__ bias,
            void* __restrict__ Out,
            int Cin, int Npad, int Nvalid, int taps, int relu, int outBf16) {
    constexpr int BM = 128, BN = 128, BK = 32;
    __shared__ __align__(16) unsigned short smA[2][BM * BK];
    __shared__ __align__(16) unsigned short smB[2][BK * BN];

    const int tid = threadIdx.x;
    const int m0  = blockIdx.x * BM;
    const int n0  = blockIdx.y * BN;
    const int ksh = (Cin == 512) ? 4 : 5;       // log2(Cin/32)
    const int kmask = (1 << ksh) - 1;
    const int NK  = taps << ksh;

    // A staging coords: 512 chunks of 8 halves; 2 per thread
    int rT[2], rb_[2], rh_[2], rw_[2];
    #pragma unroll
    for (int i = 0; i < 2; ++i) {
        int q = tid + i * 256;
        rT[i] = q >> 2;
        int rowG = m0 + rT[i];
        rb_[i] = rowG / PIX;
        int p = rowG - rb_[i] * PIX;
        rh_[i] = p / HW;
        rw_[i] = p - rh_[i] * HW;
    }
    const int kchA = tid & 3;
    // B staging coords: 512 chunks of 8 halves; 2 per thread (async)
    const int bkr0 = tid >> 4;          // 0..15
    const int bnc  = tid & 15;          // 0..15

    auto stage = [&](int ks, int buf) {
        int t  = ks >> ksh;                       // tap index (0 for 1x1)
        int c0 = (ks & kmask) << 5;               // K offset within tap
        int t3 = t / 3;
        int dy = t3 - 1, dx = t - t3 * 3 - 1;
        if (taps == 1) { dy = 0; dx = 0; }
        // --- B tile: async global -> LDS (weights, never OOB) ---
        const unsigned short* gpb =
            Wp + ((size_t)(t * Cin + c0)) * Npad + n0 + bnc * 8;
        #pragma unroll
        for (int i = 0; i < 2; ++i) {
            int kr = bkr0 + i * 16;
            const unsigned short* gp = gpb + (size_t)kr * Npad;
            unsigned lds = (unsigned)(unsigned long long)(const void*)
                           &smB[buf][kr * BN + bnc * 8];
            asm volatile("global_load_async_to_lds_b128 %0, %1, off"
                         :: "v"(lds), "v"(gp) : "memory");
        }
        // --- A tile: regular loads + ds stores, zero-fill padded rows ---
        #pragma unroll
        for (int i = 0; i < 2; ++i) {
            int hs = rh_[i] + dy, ws = rw_[i] + dx;
            uint4 v = make_uint4(0u, 0u, 0u, 0u);
            if ((unsigned)hs < (unsigned)HW && (unsigned)ws < (unsigned)HW) {
                int src = rb_[i] * PIX + hs * HW + ws;
                v = *(const uint4*)(A + (size_t)src * Cin + c0 + kchA * 8);
            }
            *(uint4*)&smA[buf][rT[i] * BK + kchA * 8] = v;
        }
    };

    // Wave tiling: 8 waves in 4(M) x 2(N); wave tile 32(M) x 64(N)
    const int lane = tid & 31;
    const int wid  = tid >> 5;
    const int wm   = (wid >> 1) * 32;
    const int wn   = (wid & 1) * 64;
    const int lr   = lane & 15;
    const int lh   = lane >> 4;

    v8f acc[2][4];
    #pragma unroll
    for (int i = 0; i < 2; ++i)
        #pragma unroll
        for (int j = 0; j < 4; ++j)
            #pragma unroll
            for (int r = 0; r < 8; ++r) acc[i][j][r] = 0.f;

    stage(0, 0);
    for (int ks = 0; ks < NK; ++ks) {
        const int buf = ks & 1;
        asm volatile("s_wait_asynccnt 0" ::: "memory");
        __syncthreads();
        if (ks + 1 < NK) stage(ks + 1, buf ^ 1);

        FragAB a[2], b[4];
        #pragma unroll
        for (int i = 0; i < 2; ++i) {
            const unsigned short* pa = &smA[buf][(wm + i * 16 + lr) * BK + lh * 8];
            a[i].h[0] = *(const v8bf*)(pa);        // K = kb .. kb+7
            a[i].h[1] = *(const v8bf*)(pa + 16);   // K = kb+16 .. kb+23
        }
        #pragma unroll
        for (int j = 0; j < 4; ++j)
            #pragma unroll
            for (int s = 0; s < 16; ++s)
                b[j].f[s] = *(const __bf16*)&smB[buf][(((s >> 3) << 4) + lh * 8 + (s & 7)) * BN + wn + j * 16 + lr]; // col N=lr, K in table order

        #pragma unroll
        for (int i = 0; i < 2; ++i)
            #pragma unroll
            for (int j = 0; j < 4; ++j)
                acc[i][j] = __builtin_amdgcn_wmma_f32_16x16x32_bf16(
                    false, a[i].f, false, b[j].f, (short)0, acc[i][j], false, false);
    }

    // Epilogue: bias (+ReLU), store bf16 NHWC or f32
    #pragma unroll
    for (int i = 0; i < 2; ++i) {
        #pragma unroll
        for (int j = 0; j < 4; ++j) {
            int nn = n0 + wn + j * 16 + lr;
            float bv = (nn < Nvalid) ? bias[nn] : 0.f;
            #pragma unroll
            for (int r = 0; r < 8; ++r) {
                int mm = m0 + wm + i * 16 + lh * 8 + r;
                float v = acc[i][j][r] + bv;
                if (relu) v = v > 0.f ? v : 0.f;
                if (outBf16)
                    ((unsigned short*)Out)[(size_t)mm * Npad + nn] = f2bf_us(v);
                else
                    ((float*)Out)[(size_t)mm * Npad + nn] = v;
            }
        }
    }
}

// ---------------------------------------------------------------------------
// logits = sigmoid(mean over pixels)   out3: f32 [M_TOT][256]
// one block per (b,nc)
// ---------------------------------------------------------------------------
__global__ void k_logits(const float* __restrict__ out3, float* __restrict__ dst) {
    __shared__ float sm[128];
    int b  = blockIdx.x / NC;
    int nc = blockIdx.x - b * NC;
    int tid = threadIdx.x;
    float s = 0.f;
    for (int p = tid; p < PIX; p += 128)
        s += out3[((size_t)(b * PIX + p)) * NPAD3 + nc];
    sm[tid] = s; __syncthreads();
    for (int st = 64; st > 0; st >>= 1) {
        if (tid < st) sm[tid] += sm[tid + st];
        __syncthreads();
    }
    if (tid == 0) {
        float v = sm[0] * (1.f / (float)PIX);
        dst[b * NC + nc] = 1.f / (1.f + __expf(-v));
    }
}

// attention map: atten[b,p] = sum_nc lab*out / max(cnt,1)
__global__ void k_atten(const float* __restrict__ out3, const int* __restrict__ label,
                        float* __restrict__ atten) {
    int m = blockIdx.x * blockDim.x + threadIdx.x;
    if (m >= M_TOT) return;
    int b = m / PIX;
    const int* lb = label + b * NC;
    const float* o = out3 + (size_t)m * NPAD3;
    float s = 0.f; int cnt = 0;
    for (int nc = 0; nc < NC; ++nc) {
        int l = lb[nc];
        if (l) { s += o[nc]; ++cnt; }
    }
    atten[m] = s / fmaxf((float)cnt, 1.f);
}

// per-batch min/max normalize + threshold -> keep mask as bf16 {0,1}
__global__ void k_normmask(const float* __restrict__ atten, unsigned short* __restrict__ keep) {
    __shared__ float smn[256], smx[256];
    int b = blockIdx.x, tid = threadIdx.x;
    const float* a = atten + b * PIX;
    float mn = 3.4e38f, mx = -3.4e38f;
    for (int p = tid; p < PIX; p += 256) { float v = a[p]; mn = fminf(mn, v); mx = fmaxf(mx, v); }
    smn[tid] = mn; smx[tid] = mx; __syncthreads();
    for (int s = 128; s > 0; s >>= 1) {
        if (tid < s) { smn[tid] = fminf(smn[tid], smn[tid + s]); smx[tid] = fmaxf(smx[tid], smx[tid + s]); }
        __syncthreads();
    }
    mn = smn[0]; mx = smx[0];
    float inv = 1.f / (mx - mn);
    for (int p = tid; p < PIX; p += 256) {
        float norm = (a[p] - mn) * inv;
        keep[b * PIX + p] = (norm >= 0.6f) ? (unsigned short)0 : (unsigned short)0x3F80; // bf16 1.0
    }
}

// featE = feat * keep (per pixel), bf16, vectorized 8 halves/thread
__global__ void k_erase(const unsigned short* __restrict__ feat,
                        const unsigned short* __restrict__ keep,
                        unsigned short* __restrict__ featE) {
    int idx = blockIdx.x * blockDim.x + threadIdx.x;   // M_TOT * 64 chunks
    if (idx >= M_TOT * (CIN0 / 8)) return;
    int m = idx >> 6;
    uint4 v = make_uint4(0u, 0u, 0u, 0u);
    if (keep[m]) v = ((const uint4*)feat)[idx];
    ((uint4*)featE)[idx] = v;
}

// ---------------------------------------------------------------------------
extern "C" void kernel_launch(void* const* d_in, const int* in_sizes, int n_in,
                              void* d_out, int out_size, void* d_ws, size_t ws_size,
                              hipStream_t stream) {
    (void)in_sizes; (void)n_in; (void)out_size; (void)ws_size;
    const float* x   = (const float*)d_in[0];
    const int* label = (const int*)d_in[1];
    const float* w1  = (const float*)d_in[2];  const float* b1 = (const float*)d_in[3];
    const float* w2  = (const float*)d_in[4];  const float* b2 = (const float*)d_in[5];
    const float* w3  = (const float*)d_in[6];  const float* b3 = (const float*)d_in[7];
    const float* we1 = (const float*)d_in[8];  const float* be1 = (const float*)d_in[9];
    const float* we2 = (const float*)d_in[10]; const float* be2 = (const float*)d_in[11];
    const float* we3 = (const float*)d_in[12]; const float* be3 = (const float*)d_in[13];
    float* outp = (float*)d_out;

    char* wsp = (char*)d_ws;
    auto carve = [&](size_t bytes) { void* p = (void*)wsp; wsp += (bytes + 255) & ~(size_t)255; return p; };
    unsigned short* feat  = (unsigned short*)carve((size_t)M_TOT * CIN0 * 2);
    unsigned short* featE = (unsigned short*)carve((size_t)M_TOT * CIN0 * 2);
    unsigned short* act1  = (unsigned short*)carve((size_t)M_TOT * 1024 * 2);
    unsigned short* act2  = (unsigned short*)carve((size_t)M_TOT * 1024 * 2);
    float*          out3  = (float*)carve((size_t)M_TOT * NPAD3 * 4);
    unsigned short* wp1   = (unsigned short*)carve((size_t)9 * 512 * 1024 * 2);
    unsigned short* wp2   = (unsigned short*)carve((size_t)9 * 1024 * 1024 * 2);
    unsigned short* wp3   = (unsigned short*)carve((size_t)1024 * NPAD3 * 2);
    unsigned short* wpe1  = (unsigned short*)carve((size_t)9 * 512 * 1024 * 2);
    unsigned short* wpe2  = (unsigned short*)carve((size_t)9 * 1024 * 1024 * 2);
    unsigned short* wpe3  = (unsigned short*)carve((size_t)1024 * NPAD3 * 2);
    float*          atten = (float*)carve((size_t)M_TOT * 4);
    unsigned short* keep  = (unsigned short*)carve((size_t)M_TOT * 2);

    // feature prep + weight packing
    k_avgpool<<<(M_TOT * CIN0 + 255) / 256, 256, 0, stream>>>(x, feat);
    k_pack3x3<<<(9 * 512 * 1024) / 256, 256, 0, stream>>>(w1, wp1, 512, 1024, 9 * 512 * 1024);
    k_pack3x3<<<(9 * 1024 * 1024) / 256, 256, 0, stream>>>(w2, wp2, 1024, 1024, 9 * 1024 * 1024);
    k_pack1x1<<<(1024 * NPAD3) / 256, 256, 0, stream>>>(w3, wp3);
    k_pack3x3<<<(9 * 512 * 1024) / 256, 256, 0, stream>>>(we1, wpe1, 512, 1024, 9 * 512 * 1024);
    k_pack3x3<<<(9 * 1024 * 1024) / 256, 256, 0, stream>>>(we2, wpe2, 1024, 1024, 9 * 1024 * 1024);
    k_pack1x1<<<(1024 * NPAD3) / 256, 256, 0, stream>>>(we3, wpe3);

    // head 1
    k_gemm<<<dim3(196, 8), 256, 0, stream>>>(feat, wp1, b1, act1, 512, 1024, 1024, 9, 1, 1);
    k_gemm<<<dim3(196, 8), 256, 0, stream>>>(act1, wp2, b2, act2, 1024, 1024, 1024, 9, 1, 1);
    k_gemm<<<dim3(196, 2), 256, 0, stream>>>(act2, wp3, b3, out3, 1024, NPAD3, NC, 1, 0, 0);
    k_logits<<<BATCH * NC, 128, 0, stream>>>(out3, outp);

    // attention + erase
    k_atten<<<(M_TOT + 255) / 256, 256, 0, stream>>>(out3, label, atten);
    k_normmask<<<BATCH, 256, 0, stream>>>(atten, keep);
    k_erase<<<(M_TOT * (CIN0 / 8)) / 256, 256, 0, stream>>>(feat, keep, featE);

    // erase head (reuses act1/act2/out3)
    k_gemm<<<dim3(196, 8), 256, 0, stream>>>(featE, wpe1, be1, act1, 512, 1024, 1024, 9, 1, 1);
    k_gemm<<<dim3(196, 8), 256, 0, stream>>>(act1, wpe2, be2, act2, 1024, 1024, 1024, 9, 1, 1);
    k_gemm<<<dim3(196, 2), 256, 0, stream>>>(act2, wpe3, be3, out3, 1024, NPAD3, NC, 1, 0, 0);
    k_logits<<<BATCH * NC, 128, 0, stream>>>(out3, outp + BATCH * NC);
}
